// TrilinearUnineTransformer_70411693851080
// MI455X (gfx1250) — matrix operand order repaired and checked by reading, not compiled, not run on hardware
//
#include <hip/hip_runtime.h>
#include <math.h>

// ---------------------------------------------------------------------------
// Types for CDNA5 WMMA (gfx1250, wave32)
// ---------------------------------------------------------------------------
typedef __bf16 bf16_t;
typedef bf16_t bf16x8  __attribute__((ext_vector_type(8)));
typedef bf16_t bf16x16 __attribute__((ext_vector_type(16)));
typedef float  f32x8   __attribute__((ext_vector_type(8)));
typedef float  f32x4   __attribute__((ext_vector_type(4)));
typedef int    i32x4   __attribute__((ext_vector_type(4)));

#define D_MODEL 768
#define FFDIM   3072
#define NHEAD   12
#define HWID    64
#define SEQ     128
#define BATCH   2
#define NTOK    256                    // B*S
#define SD      (NTOK * D_MODEL)       // 196608 elements per (B,S,D) tensor
#define NBH     (BATCH * NHEAD)        // 24
#define ATT_SCALE 0.125f               // 1/sqrt(64)
#define DD      (D_MODEL * D_MODEL)
#define DF      (D_MODEL * FFDIM)

// CDNA5 async global->LDS copy (ASYNCcnt-tracked), guarded for portability.
#if defined(__has_builtin)
#if __has_builtin(__builtin_amdgcn_global_load_async_to_lds_b128) && \
    __has_builtin(__builtin_amdgcn_s_wait_asynccnt)
#define USE_ASYNC_LDS 1
#endif
#endif
#ifndef USE_ASYNC_LDS
#define USE_ASYNC_LDS 0
#endif

#if USE_ASYNC_LDS
typedef __attribute__((address_space(1))) i32x4 as1_i32x4;   // global
typedef __attribute__((address_space(3))) i32x4 as3_i32x4;   // LDS
#endif

__device__ __forceinline__ void lds_copy16B(void* lds_dst, const void* gsrc) {
#if USE_ASYNC_LDS
    __builtin_amdgcn_global_load_async_to_lds_b128(
        (as1_i32x4*)gsrc, (as3_i32x4*)lds_dst, 0, 0);
#else
    *(bf16x8*)lds_dst = *(const bf16x8*)gsrc;
#endif
}

__device__ __forceinline__ void lds_copy_wait() {
#if USE_ASYNC_LDS
    __builtin_amdgcn_s_wait_asynccnt(0);
#endif
}

__device__ __forceinline__ f32x8 wmma_bf16(bf16x16 a, bf16x16 b, f32x8 c) {
    // v_wmma_f32_16x16x32_bf16  D = A x B + C
    return __builtin_amdgcn_wmma_f32_16x16x32_bf16(
        false, a, false, b, (short)0, c, false, false);
}

// A fragment (16x32 bf16, M x K): lane = m | 16*kh ; element e<8 -> K=kh*8+e,
// e>=8 -> K=16+kh*8+(e-8)  (ISA 7.12.2 "16-bit A-Matrix 16x32")
__device__ __forceinline__ bf16x16 load_a_frag(const bf16_t* Abuf, int lda,
                                               int mBase, int kBase, int lane) {
    int m  = mBase + (lane & 15);
    int kh = (lane >> 4) * 8;
    bf16x8 lo = *(const bf16x8*)(Abuf + m * lda + kBase + kh);
    bf16x8 hi = *(const bf16x8*)(Abuf + m * lda + kBase + 16 + kh);
    bf16x16 r;
#pragma unroll
    for (int i = 0; i < 8; ++i) { r[i] = lo[i]; r[i + 8] = hi[i]; }
    return r;
}

// B fragment (32x16 bf16, K x N) from an N-major LDS tile Bt[n][k]:
// lane = n | 16*kh ; element e<8 -> K=kh*8+e, e>=8 -> K=16+kh*8+(e-8)
__device__ __forceinline__ bf16x16 load_b_frag(const bf16_t* Bbuf, int ldb,
                                               int nBase, int kBase, int lane) {
    int n  = nBase + (lane & 15);
    int kh = (lane >> 4) * 8;
    bf16x8 lo = *(const bf16x8*)(Bbuf + n * ldb + kBase + kh);
    bf16x8 hi = *(const bf16x8*)(Bbuf + n * ldb + kBase + 16 + kh);
    bf16x16 r;
#pragma unroll
    for (int i = 0; i < 8; ++i) { r[i] = lo[i]; r[i + 8] = hi[i]; }
    return r;
}

__device__ __forceinline__ float gelu_exact(float x) {
    return 0.5f * x * (1.0f + erff(x * 0.70710678118654752f));
}

// ---------------------------------------------------------------------------
// Weight pre-pass: out[n][k] = (bf16) in[k][n]  (32x32 LDS-tiled transpose)
// ---------------------------------------------------------------------------
__global__ __launch_bounds__(256) void wconv_kernel(
    const float* __restrict__ in, bf16_t* __restrict__ outp, int K, int N)
{
    __shared__ float tile[32][33];
    const int kb = blockIdx.x * 32;
    const int nb = blockIdx.y * 32;
    const int tx = threadIdx.x & 31;
    const int ty = threadIdx.x >> 5;   // 0..7
#pragma unroll
    for (int r = 0; r < 32; r += 8)
        tile[ty + r][tx] = in[(size_t)(kb + ty + r) * N + nb + tx];
    __syncthreads();
#pragma unroll
    for (int r = 0; r < 32; r += 8)
        outp[(size_t)(nb + ty + r) * K + kb + tx] = (bf16_t)tile[tx][ty + r];
}

// ---------------------------------------------------------------------------
// bf16-WMMA GEMM: C[M,N] = epilogue(A[M,K] @ W[K,N] + bias [+ res])
// A fp32 (converted per tile), Wt = bf16 pre-transposed [N][K].
// BM=128, BN=128, BK=32; 256 threads = 8 waves (2x4), each wave 64x32 output.
// act: 0 = none, 1 = exact GELU.
// ---------------------------------------------------------------------------
#define GM_BM 128
#define GM_BN 128
#define GM_BK 32

__global__ __launch_bounds__(256) void gemm_bf16_kernel(
    const float* __restrict__ A, const bf16_t* __restrict__ Wt,
    const float* __restrict__ bias, const float* __restrict__ res,
    float* __restrict__ C, int M, int N, int K, int act)
{
    __shared__ bf16_t As[GM_BM * GM_BK];   // row-major [m][k]
    __shared__ bf16_t Bs[GM_BN * GM_BK];   // N-major   [n][k]

    const int tid  = threadIdx.x;
    const int lane = tid & 31;
    const int wave = tid >> 5;
    const int wm   = (wave >> 2) * 64;     // wave row strip (2 strips)
    const int wn   = (wave & 3) * 32;      // wave col strip (4 strips)
    const int m0   = blockIdx.x * GM_BM;
    const int n0   = blockIdx.y * GM_BN;

    f32x8 acc[4][2];
#pragma unroll
    for (int i = 0; i < 4; ++i)
#pragma unroll
        for (int j = 0; j < 2; ++j)
#pragma unroll
            for (int r = 0; r < 8; ++r) acc[i][j][r] = 0.0f;

    for (int k0 = 0; k0 < K; k0 += GM_BK) {
        // ---- stage A tile (fp32 -> bf16 convert) ----
        {
            int row = tid >> 1;
            int seg = (tid & 1) * 16;
            const float* src = A + (size_t)(m0 + row) * K + k0 + seg;
            bf16_t* dst = As + row * GM_BK + seg;
#pragma unroll
            for (int i = 0; i < 16; i += 4) {
                f32x4 v = *(const f32x4*)(src + i);
                dst[i + 0] = (bf16_t)v.x; dst[i + 1] = (bf16_t)v.y;
                dst[i + 2] = (bf16_t)v.z; dst[i + 3] = (bf16_t)v.w;
            }
            if (k0 + GM_BK < K) __builtin_prefetch(src + GM_BK, 0, 1);
        }
        // ---- stage B tile: raw bf16 16B chunks (async global->LDS) ----
        {
            int n   = tid >> 1;
            int off = (tid & 1) * 16;
            const bf16_t* gsrc = Wt + (size_t)(n0 + n) * K + k0 + off;
            bf16_t* ldst = Bs + n * GM_BK + off;
            lds_copy16B(ldst, gsrc);
            lds_copy16B(ldst + 8, gsrc + 8);
        }
        lds_copy_wait();
        __syncthreads();

        bf16x16 af[4], bfr[2];
#pragma unroll
        for (int i = 0; i < 4; ++i)
            af[i] = load_a_frag(As, GM_BK, wm + i * 16, 0, lane);
#pragma unroll
        for (int j = 0; j < 2; ++j)
            bfr[j] = load_b_frag(Bs, GM_BK, wn + j * 16, 0, lane);
#pragma unroll
        for (int i = 0; i < 4; ++i)
#pragma unroll
            for (int j = 0; j < 2; ++j)
                acc[i][j] = wmma_bf16(af[i], bfr[j], acc[i][j]);
        __syncthreads();
    }

    // ---- epilogue: bias, optional GELU, optional residual ----
    const int nlane = lane & 15;
    const int mh    = (lane >> 4) * 8;
#pragma unroll
    for (int i = 0; i < 4; ++i) {
#pragma unroll
        for (int j = 0; j < 2; ++j) {
            int gmB = m0 + wm + i * 16 + mh;
            int gn  = n0 + wn + j * 16 + nlane;
            float bv = bias ? bias[gn] : 0.0f;
#pragma unroll
            for (int r = 0; r < 8; ++r) {
                float x = acc[i][j][r] + bv;
                if (act == 1) x = gelu_exact(x);
                size_t idx = (size_t)(gmB + r) * N + gn;
                if (res) x += res[idx];
                C[idx] = x;
            }
        }
    }
}

// ---------------------------------------------------------------------------
// LayerNorm over last dim (768). One block per token row.
// ---------------------------------------------------------------------------
__global__ __launch_bounds__(256) void ln_kernel(
    const float* __restrict__ X, const float* __restrict__ g,
    const float* __restrict__ b, float* __restrict__ Y)
{
    __shared__ float red[256];
    const int tid = threadIdx.x;
    const float* x = X + (size_t)blockIdx.x * D_MODEL;
    float s = 0.0f;
    for (int i = tid; i < D_MODEL; i += 256) s += x[i];
    red[tid] = s; __syncthreads();
    for (int k = 128; k > 0; k >>= 1) {
        if (tid < k) red[tid] += red[tid + k];
        __syncthreads();
    }
    float mean = red[0] / (float)D_MODEL;
    __syncthreads();
    float v = 0.0f;
    for (int i = tid; i < D_MODEL; i += 256) { float d = x[i] - mean; v += d * d; }
    red[tid] = v; __syncthreads();
    for (int k = 128; k > 0; k >>= 1) {
        if (tid < k) red[tid] += red[tid + k];
        __syncthreads();
    }
    float rstd = rsqrtf(red[0] / (float)D_MODEL + 1e-6f);
    float* y = Y + (size_t)blockIdx.x * D_MODEL;
    for (int i = tid; i < D_MODEL; i += 256)
        y[i] = (x[i] - mean) * rstd * g[i] + b[i];
}

// ---------------------------------------------------------------------------
// Trilinear attention. One block per (b,h,v): computes the 128x128 (q,a) slice
// att[v,:,:] = (hv[v]*hq) @ ha^T * SCALE via bf16 WMMA (M=128,N=128,K=64).
// ---------------------------------------------------------------------------
union TriSmem {
    struct {
        bf16_t Aq[128 * 64];   // G[q][w] = hv[v][w]*hq[q][w]
        bf16_t Ba[128 * 64];   // ha[a][w] (already B^T in N-major layout)
        float  hv[64];
    } s;
    float red[256];
    struct {
        float rowsum[128];     // over q
        float colsum[128];     // over a
        float total;
    } m;
};

__device__ void tri_compute_tile(const float* __restrict__ pv,
                                 const float* __restrict__ pq,
                                 const float* __restrict__ pa,
                                 int b, int h, int vi, int tid,
                                 TriSmem* sm, f32x8 acc[2][4])
{
    if (tid < 16) {
        f32x4 t = *(const f32x4*)(pv + (size_t)(b * SEQ + vi) * D_MODEL + h * HWID + tid * 4);
        sm->s.hv[tid * 4 + 0] = t.x; sm->s.hv[tid * 4 + 1] = t.y;
        sm->s.hv[tid * 4 + 2] = t.z; sm->s.hv[tid * 4 + 3] = t.w;
    }
    __syncthreads();

    const int row = tid >> 1;
    const int seg = (tid & 1) * 32;
    const float* qrow = pq + (size_t)(b * SEQ + row) * D_MODEL + h * HWID + seg;
    const float* arow = pa + (size_t)(b * SEQ + row) * D_MODEL + h * HWID + seg;
#pragma unroll
    for (int i = 0; i < 32; i += 4) {
        f32x4 qv = *(const f32x4*)(qrow + i);
        f32x4 av = *(const f32x4*)(arow + i);
        int w = seg + i;
        sm->s.Aq[row * 64 + w + 0] = (bf16_t)(qv.x * sm->s.hv[w + 0]);
        sm->s.Aq[row * 64 + w + 1] = (bf16_t)(qv.y * sm->s.hv[w + 1]);
        sm->s.Aq[row * 64 + w + 2] = (bf16_t)(qv.z * sm->s.hv[w + 2]);
        sm->s.Aq[row * 64 + w + 3] = (bf16_t)(qv.w * sm->s.hv[w + 3]);
        sm->s.Ba[row * 64 + w + 0] = (bf16_t)av.x;
        sm->s.Ba[row * 64 + w + 1] = (bf16_t)av.y;
        sm->s.Ba[row * 64 + w + 2] = (bf16_t)av.z;
        sm->s.Ba[row * 64 + w + 3] = (bf16_t)av.w;
    }
    __syncthreads();

    const int lane = tid & 31;
    const int wave = tid >> 5;
    const int wm = (wave >> 1) * 32;   // q strip
    const int wn = (wave & 1) * 64;    // a strip
#pragma unroll
    for (int kb = 0; kb < 64; kb += 32) {
        bf16x16 af0 = load_a_frag(sm->s.Aq, 64, wm,      kb, lane);
        bf16x16 af1 = load_a_frag(sm->s.Aq, 64, wm + 16, kb, lane);
#pragma unroll
        for (int j = 0; j < 4; ++j) {
            bf16x16 bfj = load_b_frag(sm->s.Ba, 64, wn + j * 16, kb, lane);
            acc[0][j] = wmma_bf16(af0, bfj, acc[0][j]);
            acc[1][j] = wmma_bf16(af1, bfj, acc[1][j]);
        }
    }
}

// Pass 1: per-block (fixed v) online-softmax partials (max, sum-exp).
__global__ __launch_bounds__(256) void tri_pass1_kernel(
    const float* __restrict__ pv, const float* __restrict__ pq,
    const float* __restrict__ pa, float* __restrict__ mpart,
    float* __restrict__ spart)
{
    __shared__ TriSmem sm;
    const int tid = threadIdx.x;
    const int blk = blockIdx.x;
    const int vi  = blk & (SEQ - 1);
    const int bh  = blk >> 7;
    const int b = bh / NHEAD, h = bh % NHEAD;

    f32x8 acc[2][4];
#pragma unroll
    for (int i = 0; i < 2; ++i)
#pragma unroll
        for (int j = 0; j < 4; ++j)
#pragma unroll
            for (int r = 0; r < 8; ++r) acc[i][j][r] = 0.0f;

    tri_compute_tile(pv, pq, pa, b, h, vi, tid, &sm, acc);

    float lm = -INFINITY;
#pragma unroll
    for (int i = 0; i < 2; ++i)
#pragma unroll
        for (int j = 0; j < 4; ++j)
#pragma unroll
            for (int r = 0; r < 8; ++r)
                lm = fmaxf(lm, acc[i][j][r] * ATT_SCALE);

    __syncthreads();                 // done reading sm.s
    sm.red[tid] = lm; __syncthreads();
    for (int k = 128; k > 0; k >>= 1) {
        if (tid < k) sm.red[tid] = fmaxf(sm.red[tid], sm.red[tid + k]);
        __syncthreads();
    }
    float bm = sm.red[0];
    __syncthreads();

    float ls = 0.0f;
#pragma unroll
    for (int i = 0; i < 2; ++i)
#pragma unroll
        for (int j = 0; j < 4; ++j)
#pragma unroll
            for (int r = 0; r < 8; ++r)
                ls += __expf(acc[i][j][r] * ATT_SCALE - bm);
    sm.red[tid] = ls; __syncthreads();
    for (int k = 128; k > 0; k >>= 1) {
        if (tid < k) sm.red[tid] += sm.red[tid + k];
        __syncthreads();
    }
    if (tid == 0) { mpart[blk] = bm; spart[blk] = sm.red[0]; }
}

// Combine 128 per-v partials into global (M, Z) per (b,h).
__global__ __launch_bounds__(128) void tri_reduce_kernel(
    const float* __restrict__ mpart, const float* __restrict__ spart,
    float* __restrict__ Mz)
{
    __shared__ float rm[128], rs[128];
    const int tid = threadIdx.x;
    const int bh  = blockIdx.x;
    float m = mpart[bh * SEQ + tid];
    float s = spart[bh * SEQ + tid];
    rm[tid] = m; __syncthreads();
    for (int k = 64; k > 0; k >>= 1) {
        if (tid < k) rm[tid] = fmaxf(rm[tid], rm[tid + k]);
        __syncthreads();
    }
    float M = rm[0];
    rs[tid] = s * __expf(m - M); __syncthreads();
    for (int k = 64; k > 0; k >>= 1) {
        if (tid < k) rs[tid] += rs[tid + k];
        __syncthreads();
    }
    if (tid == 0) { Mz[bh * 2] = M; Mz[bh * 2 + 1] = rs[0]; }
}

// Pass 2: recompute tile, p = exp(x-M)/Z, accumulate the three marginals.
__global__ __launch_bounds__(256) void tri_pass2_kernel(
    const float* __restrict__ pv, const float* __restrict__ pq,
    const float* __restrict__ pa, const float* __restrict__ Mz,
    float* __restrict__ Sv, float* __restrict__ Sq, float* __restrict__ Sa)
{
    __shared__ TriSmem sm;
    const int tid = threadIdx.x;
    const int blk = blockIdx.x;
    const int vi  = blk & (SEQ - 1);
    const int bh  = blk >> 7;
    const int b = bh / NHEAD, h = bh % NHEAD;

    f32x8 acc[2][4];
#pragma unroll
    for (int i = 0; i < 2; ++i)
#pragma unroll
        for (int j = 0; j < 4; ++j)
#pragma unroll
            for (int r = 0; r < 8; ++r) acc[i][j][r] = 0.0f;

    tri_compute_tile(pv, pq, pa, b, h, vi, tid, &sm, acc);

    const float M  = Mz[bh * 2];
    const float rZ = 1.0f / Mz[bh * 2 + 1];

    __syncthreads();                 // done reading sm.s
    if (tid < 128) { sm.m.rowsum[tid] = 0.0f; sm.m.colsum[tid] = 0.0f; }
    if (tid == 0) sm.m.total = 0.0f;
    __syncthreads();

    const int lane = tid & 31;
    const int wave = tid >> 5;
    const int wm = (wave >> 1) * 32;
    const int wn = (wave & 1) * 64;
    const int nlane = lane & 15;
    const int mh = (lane >> 4) * 8;

    float lt = 0.0f;
#pragma unroll
    for (int i = 0; i < 2; ++i) {
#pragma unroll
        for (int j = 0; j < 4; ++j) {
#pragma unroll
            for (int r = 0; r < 8; ++r) {
                float p = __expf(acc[i][j][r] * ATT_SCALE - M) * rZ;
                int qi = wm + i * 16 + mh + r;
                int ai = wn + j * 16 + nlane;
                atomicAdd(&sm.m.rowsum[qi], p);
                atomicAdd(&sm.m.colsum[ai], p);
                lt += p;
            }
        }
    }
    atomicAdd(&sm.m.total, lt);
    __syncthreads();
    if (tid < 128) {
        atomicAdd(&Sq[bh * SEQ + tid], sm.m.rowsum[tid]);
        atomicAdd(&Sa[bh * SEQ + tid], sm.m.colsum[tid]);
    }
    if (tid == 0) Sv[bh * SEQ + vi] = sm.m.total;  // block covers all (q,a) for this v
}

// out[b,s,d] = src[b,s,d] * (per_s ? marg[b,h(d),s] : marg[b,h(d)])
__global__ __launch_bounds__(256) void apply_marg_kernel(
    const float* __restrict__ src, const float* __restrict__ marg,
    float* __restrict__ dst, int per_s)
{
    int i = blockIdx.x * 256 + threadIdx.x;
    if (i >= SD) return;
    int d = i % D_MODEL;
    int s = (i / D_MODEL) % SEQ;
    int b = i / (D_MODEL * SEQ);
    int bh = b * NHEAD + (d >> 6);
    float sc = per_s ? marg[bh * SEQ + s] : marg[bh];
    dst[i] = src[i] * sc;
}

// _mha_al stats: closed-form marginals of the multilinear (un-softmaxed) att1.
__global__ __launch_bounds__(128) void al_stats_kernel(
    const float* __restrict__ pv, const float* __restrict__ pq,
    const float* __restrict__ pk, float* __restrict__ rowv,
    float* __restrict__ rowq, float* __restrict__ Tout)
{
    __shared__ float Hv[64], Hq[64], Hk[64], red[128];
    const int tid = threadIdx.x;
    const int bh  = blockIdx.x;
    const int b = bh / NHEAD, h = bh % NHEAD;

    if (tid < 64) {
        float sv = 0.0f, sq = 0.0f, sk = 0.0f;
        for (int r = 0; r < SEQ; ++r) {
            size_t off = (size_t)(b * SEQ + r) * D_MODEL + h * HWID + tid;
            sv += pv[off]; sq += pq[off]; sk += pk[off];
        }
        Hv[tid] = sv; Hq[tid] = sq; Hk[tid] = sk;
    }
    __syncthreads();

    float t = (tid < 64) ? Hv[tid] * Hq[tid] * Hk[tid] : 0.0f;
    red[tid] = t; __syncthreads();
    for (int k = 64; k > 0; k >>= 1) {
        if (tid < k) red[tid] += red[tid + k];
        __syncthreads();
    }
    if (tid == 0) Tout[bh] = red[0] * ATT_SCALE;

    const float* vrow = pv + (size_t)(b * SEQ + tid) * D_MODEL + h * HWID;
    const float* qrow = pq + (size_t)(b * SEQ + tid) * D_MODEL + h * HWID;
    float accv = 0.0f, accq = 0.0f;
#pragma unroll 8
    for (int w = 0; w < 64; ++w) {
        accv += vrow[w] * Hq[w] * Hk[w];
        accq += Hv[w] * qrow[w] * Hk[w];
    }
    rowv[bh * SEQ + tid] = accv * ATT_SCALE;
    rowq[bh * SEQ + tid] = accq * ATT_SCALE;
}

__global__ __launch_bounds__(256) void zero_kernel(float* __restrict__ p, int n) {
    int i = blockIdx.x * 256 + threadIdx.x;
    if (i < n) p[i] = 0.0f;
}

// ---------------------------------------------------------------------------
// Host orchestration
// ---------------------------------------------------------------------------
extern "C" void kernel_launch(void* const* d_in, const int* in_sizes, int n_in,
                              void* d_out, int out_size, void* d_ws, size_t ws_size,
                              hipStream_t stream) {
    const float* in_v      = (const float*)d_in[0];
    const float* in_q      = (const float*)d_in[1];
    const float* in_a      = (const float*)d_in[2];
    const float* in_k      = (const float*)d_in[3];
    const float* tri_Win   = (const float*)d_in[4];
    const float* tri_bin   = (const float*)d_in[5];
    const float* tri_Wout  = (const float*)d_in[6];
    const float* tri_bout  = (const float*)d_in[7];
    const float* tri_Wp    = (const float*)d_in[8];
    const float* tri_bp    = (const float*)d_in[9];
    const float* tri_g1    = (const float*)d_in[10];
    const float* tri_beta1 = (const float*)d_in[11];
    const float* tri_g2    = (const float*)d_in[12];
    const float* tri_beta2 = (const float*)d_in[13];
    const float* tri_W1    = (const float*)d_in[14];
    const float* tri_b1    = (const float*)d_in[15];
    const float* tri_W2    = (const float*)d_in[16];
    const float* tri_b2    = (const float*)d_in[17];
    const float* al_Win    = (const float*)d_in[18];
    const float* al_bin    = (const float*)d_in[19];
    const float* al_Wout   = (const float*)d_in[20];
    const float* al_bout   = (const float*)d_in[21];
    const float* out_W     = (const float*)d_in[22];
    const float* out_b     = (const float*)d_in[23];
    float* out = (float*)d_out;

    float* ws = (float*)d_ws;
    float* v   = ws + 0  * (size_t)SD;
    float* q   = ws + 1  * (size_t)SD;
    float* a   = ws + 2  * (size_t)SD;
    float* nv  = ws + 3  * (size_t)SD;   // LN output / generic temp
    float* nq  = ws + 4  * (size_t)SD;
    float* na  = ws + 5  * (size_t)SD;
    float* pv  = ws + 6  * (size_t)SD;
    float* pq  = ws + 7  * (size_t)SD;
    float* pa  = ws + 8  * (size_t)SD;
    float* pk  = ws + 9  * (size_t)SD;
    float* vo  = ws + 10 * (size_t)SD;
    float* qo  = ws + 11 * (size_t)SD;
    float* ao  = ws + 12 * (size_t)SD;
    float* ko  = ws + 13 * (size_t)SD;
    float* ff1 = ws + 14 * (size_t)SD;   // 256 x 3072 = 4*SD
    float* st  = ws + 18 * (size_t)SD;
    float* mpart = st;                   // 3072
    float* spart = mpart + NBH * SEQ;    // 3072
    float* Mz    = spart + NBH * SEQ;    // 48
    float* Sv    = Mz + 2 * NBH;         // 3072
    float* Sq    = Sv + NBH * SEQ;
    float* Sa    = Sq + NBH * SEQ;
    float* rowv  = Sa + NBH * SEQ;
    float* rowq  = rowv + NBH * SEQ;
    float* Tt    = rowq + NBH * SEQ;     // 24
    bf16_t* wt   = (bf16_t*)(ws + 19 * (size_t)SD);  // DF bf16 = 6*SD floats

    (void)hipMemcpyAsync(v, in_v, (size_t)SD * 4, hipMemcpyDeviceToDevice, stream);
    (void)hipMemcpyAsync(q, in_q, (size_t)SD * 4, hipMemcpyDeviceToDevice, stream);
    (void)hipMemcpyAsync(a, in_a, (size_t)SD * 4, hipMemcpyDeviceToDevice, stream);

    // pre-transpose+convert one weight into wt (bf16 [N][K])
    auto conv = [&](const float* W, int K, int N) {
        dim3 g(K / 32, N / 32);
        wconv_kernel<<<g, 256, 0, stream>>>(W, wt, K, N);
    };
    // GEMM using the already-staged wt
    auto gemm_pre = [&](const float* A, const float* bias, const float* res,
                        float* C, int M, int N, int K, int act) {
        dim3 grid(M / GM_BM, N / GM_BN);
        gemm_bf16_kernel<<<grid, 256, 0, stream>>>(A, wt, bias, res, C, M, N, K, act);
    };
    auto gemm = [&](const float* A, const float* W, const float* bias,
                    const float* res, float* C, int M, int N, int K, int act) {
        conv(W, K, N);
        gemm_pre(A, bias, res, C, M, N, K, act);
    };
    auto ln = [&](const float* X, const float* g, const float* b, float* Y) {
        ln_kernel<<<NTOK, 256, 0, stream>>>(X, g, b, Y);
    };

    for (int l = 0; l < 2; ++l) {
        const float* Win  = tri_Win  + (size_t)l * 3 * DD;
        const float* bin  = tri_bin  + (size_t)l * 3 * D_MODEL;
        const float* Wout = tri_Wout + (size_t)l * 3 * DD;
        const float* bout = tri_bout + (size_t)l * 3 * D_MODEL;
        const float* Wp = tri_Wp + (size_t)l * DD;
        const float* bp = tri_bp + (size_t)l * D_MODEL;
        const float* g1 = tri_g1 + (size_t)l * D_MODEL;
        const float* be1 = tri_beta1 + (size_t)l * D_MODEL;
        const float* g2 = tri_g2 + (size_t)l * D_MODEL;
        const float* be2 = tri_beta2 + (size_t)l * D_MODEL;
        const float* W1 = tri_W1 + (size_t)l * DF;
        const float* b1 = tri_b1 + (size_t)l * FFDIM;
        const float* W2 = tri_W2 + (size_t)l * DF;
        const float* b2 = tri_b2 + (size_t)l * D_MODEL;

        // LayerNorm of the three streams (pre-update values)
        ln(v, g1, be1, nv); ln(q, g1, be1, nq); ln(a, g1, be1, na);
        // in-projections
        gemm(nv, Win + 0 * DD, bin + 0 * D_MODEL, nullptr, pv, NTOK, D_MODEL, D_MODEL, 0);
        gemm(nq, Win + 1 * DD, bin + 1 * D_MODEL, nullptr, pq, NTOK, D_MODEL, D_MODEL, 0);
        gemm(na, Win + 2 * DD, bin + 2 * D_MODEL, nullptr, pa, NTOK, D_MODEL, D_MODEL, 0);
        // out-projections (applied to projected streams, per reference)
        gemm(pv, Wout + 0 * DD, bout + 0 * D_MODEL, nullptr, vo, NTOK, D_MODEL, D_MODEL, 0);
        gemm(pq, Wout + 1 * DD, bout + 1 * D_MODEL, nullptr, qo, NTOK, D_MODEL, D_MODEL, 0);
        gemm(pa, Wout + 2 * DD, bout + 2 * D_MODEL, nullptr, ao, NTOK, D_MODEL, D_MODEL, 0);

        // trilinear attention: softmax stats then marginals
        tri_pass1_kernel<<<NBH * SEQ, 256, 0, stream>>>(pv, pq, pa, mpart, spart);
        tri_reduce_kernel<<<NBH, 128, 0, stream>>>(mpart, spart, Mz);
        zero_kernel<<<(2 * NBH * SEQ + 255) / 256, 256, 0, stream>>>(Sq, 2 * NBH * SEQ);
        tri_pass2_kernel<<<NBH * SEQ, 256, 0, stream>>>(pv, pq, pa, Mz, Sv, Sq, Sa);

        // apply marginals, project, residual add (in place on streams)
        conv(Wp, D_MODEL, D_MODEL);
        apply_marg_kernel<<<SD / 256, 256, 0, stream>>>(vo, Sv, nv, 1);
        gemm_pre(nv, bp, v, v, NTOK, D_MODEL, D_MODEL, 0);
        apply_marg_kernel<<<SD / 256, 256, 0, stream>>>(qo, Sq, nv, 1);
        gemm_pre(nv, bp, q, q, NTOK, D_MODEL, D_MODEL, 0);
        apply_marg_kernel<<<SD / 256, 256, 0, stream>>>(ao, Sa, nv, 1);
        gemm_pre(nv, bp, a, a, NTOK, D_MODEL, D_MODEL, 0);

        // FFN per stream: x += GELU(LN(x)@W1+b1)@W2+b2
        float* streams[3] = { v, q, a };
        for (int sidx = 0; sidx < 3; ++sidx) {
            float* x = streams[sidx];
            ln(x, g2, be2, nv);
            gemm(nv, W1, b1, nullptr, ff1, NTOK, FFDIM, D_MODEL, 1);
            gemm(ff1, W2, b2, x, x, NTOK, D_MODEL, FFDIM, 0);
        }
    }

    // ---- _mha_al (no LN, raw streams + k input) ----
    gemm(v,    al_Win + 0 * DD, al_bin + 0 * D_MODEL, nullptr, pv, NTOK, D_MODEL, D_MODEL, 0);
    gemm(q,    al_Win + 1 * DD, al_bin + 1 * D_MODEL, nullptr, pq, NTOK, D_MODEL, D_MODEL, 0);
    gemm(a,    al_Win + 2 * DD, al_bin + 2 * D_MODEL, nullptr, pa, NTOK, D_MODEL, D_MODEL, 0);
    gemm(in_k, al_Win + 3 * DD, al_bin + 3 * D_MODEL, nullptr, pk, NTOK, D_MODEL, D_MODEL, 0);
    gemm(pv, al_Wout + 0 * DD, al_bout + 0 * D_MODEL, nullptr, vo, NTOK, D_MODEL, D_MODEL, 0);
    gemm(pq, al_Wout + 1 * DD, al_bout + 1 * D_MODEL, nullptr, qo, NTOK, D_MODEL, D_MODEL, 0);
    gemm(pa, al_Wout + 2 * DD, al_bout + 2 * D_MODEL, nullptr, ao, NTOK, D_MODEL, D_MODEL, 0);
    gemm(pk, al_Wout + 3 * DD, al_bout + 3 * D_MODEL, nullptr, ko, NTOK, D_MODEL, D_MODEL, 0);

    al_stats_kernel<<<NBH, 128, 0, stream>>>(pv, pq, pk, rowv, rowq, Tt);

    // final output projection into d_out (4 tensors, concatenated).
    // out_W is reused 4x: convert once, keep wt resident.
    conv(out_W, D_MODEL, D_MODEL);
    apply_marg_kernel<<<SD / 256, 256, 0, stream>>>(vo, rowv, nv, 1);
    gemm_pre(nv, out_b, nullptr, out + 0 * (size_t)SD, NTOK, D_MODEL, D_MODEL, 0);
    apply_marg_kernel<<<SD / 256, 256, 0, stream>>>(qo, rowq, nq, 1);
    gemm_pre(nq, out_b, nullptr, out + 1 * (size_t)SD, NTOK, D_MODEL, D_MODEL, 0);
    apply_marg_kernel<<<SD / 256, 256, 0, stream>>>(ao, Tt, na, 0);
    gemm_pre(na, out_b, nullptr, out + 2 * (size_t)SD, NTOK, D_MODEL, D_MODEL, 0);
    gemm_pre(ko, out_b, nullptr, out + 3 * (size_t)SD, NTOK, D_MODEL, D_MODEL, 0);
}
